// DGCNN_17849884082381
// MI455X (gfx1250) — compile-verified, matrix operand order checked by reading, not run on hardware
//
#include <hip/hip_runtime.h>

#define NPTS   2048
#define BATCH  8
#define KNN    20
#define FEATC  512

typedef __attribute__((ext_vector_type(16))) _Float16 v16h;
typedef __attribute__((ext_vector_type(8)))  float    v8f;
typedef __attribute__((ext_vector_type(4)))  unsigned u32x4;
typedef __attribute__((ext_vector_type(8)))  int      i32x8;
typedef __attribute__((ext_vector_type(4)))  int      i32x4;

// ---------------------------------------------------------------------------
// Tensor Data Mover: 2-D tile load (f16 elements) global -> LDS.
// D# packing per CDNA5 ISA ch.8: group0 {count=1 | lds_addr | gaddr57 | type=2},
// group1 {wg_mask=0, data_size=1(2B), dims/strides}, groups 2/3 zero (2-D).
// 6-arg builtin form (clang-23 / therock-10.0 headers).
// ---------------------------------------------------------------------------
__device__ __forceinline__ void tdm_load_2d(unsigned lds_byte_off, const void* gptr,
                                            unsigned tensor_d0, unsigned tensor_d1,
                                            unsigned tile_d0, unsigned tile_d1,
                                            unsigned d0_stride_elems) {
  unsigned long long ga = (unsigned long long)(uintptr_t)gptr;
  u32x4 g0;
  g0[0] = 1u;                                                  // count=1 (valid user D#)
  g0[1] = lds_byte_off;                                        // lds_addr (bytes)
  g0[2] = (unsigned)ga;                                        // global_addr[31:0]
  g0[3] = (unsigned)((ga >> 32) & 0x01FFFFFFu) | (2u << 30);   // addr[56:32] | type=2
  i32x8 g1;
  g1[0] = (int)(1u << 16);                                     // data_size=1 -> 2 bytes
  g1[1] = (int)((tensor_d0 & 0xFFFFu) << 16);                  // tensor_dim0[15:0]
  g1[2] = (int)((tensor_d0 >> 16) | ((tensor_d1 & 0xFFFFu) << 16));
  g1[3] = (int)((tensor_d1 >> 16) | (tile_d0 << 16));          // tile_dim0
  g1[4] = (int)(tile_d1 & 0xFFFFu);                            // tile_dim1 (tile_dim2=0)
  g1[5] = (int)d0_stride_elems;                                // tensor_dim0_stride[31:0]
  g1[6] = 0;
  g1[7] = 0;
  i32x4 z4 = {};
  i32x8 z8 = {};
  __builtin_amdgcn_tensor_load_to_lds(g0, g1, z4, z4, z8, 0);
}

// ---------------------------------------------------------------------------
// WMMA fragment loaders. LDS tiles are [c][16 points] (row = 32B, TDM order).
// A (16x32 f16, global row-major): lane holds row m=L&15; vgpr v holds K pair
//   k0 = (v>>2)*16 + (L>>4)*8 + (v&3)*2
// B (32x16 f16 from LDS): lane holds column n=L&15; same K mapping.
// C/D (16x16 f32): lane L, vgpr v -> (m = v + 8*(L>>4), n = L&15)
// ---------------------------------------------------------------------------
__device__ __forceinline__ v16h load_a_frag(const _Float16* __restrict__ W,
                                            int ldw, int rowbase, int kbase) {
  int lane = threadIdx.x & 31;
  const _Float16* p = W + (size_t)(rowbase + (lane & 15)) * ldw + kbase + ((lane >> 4) << 3);
  v16h a;
#pragma unroll
  for (int v = 0; v < 8; ++v) {
    int k0 = ((v >> 2) << 4) + ((v & 3) << 1);
    a[2 * v]     = p[k0];
    a[2 * v + 1] = p[k0 + 1];
  }
  return a;
}

__device__ __forceinline__ v16h load_b_frag16(const _Float16* __restrict__ tile, int kbase) {
  int lane = threadIdx.x & 31;
  const _Float16* p = tile + ((kbase + ((lane >> 4) << 3)) << 4) + (lane & 15);
  v16h b;
#pragma unroll
  for (int v = 0; v < 8; ++v) {
    int k0 = ((v >> 2) << 4) + ((v & 3) << 1);
    b[2 * v]     = p[k0 << 4];
    b[2 * v + 1] = p[(k0 + 1) << 4];
  }
  return b;
}

// ordered-uint encoding for float atomic max
__device__ __forceinline__ unsigned encf(float f) {
  unsigned u = __float_as_uint(f);
  return (u & 0x80000000u) ? ~u : (u | 0x80000000u);
}
__device__ __forceinline__ float decf(unsigned e) {
  unsigned u = (e & 0x80000000u) ? (e ^ 0x80000000u) : ~e;
  return __uint_as_float(u);
}

// ---------------------------------------------------------------------------
// small utility kernels
// ---------------------------------------------------------------------------
__global__ void zero_kernel(float* p, int n) {
  int i = blockIdx.x * 256 + threadIdx.x;
  if (i < n) p[i] = 0.f;
}

__global__ void init_enc_kernel(unsigned* p, int n) {
  int i = blockIdx.x * 256 + threadIdx.x;
  if (i < n) p[i] = encf(-__builtin_inff());
}

__global__ void decode_kernel(const unsigned* __restrict__ e, float* __restrict__ out, int n) {
  int i = blockIdx.x * 256 + threadIdx.x;
  if (i < n) out[i] = decf(e[i]);
}

// x (B,N,3) -> xt (B,3,N) fp32 + f16 mirror (TDM source)
__global__ void transpose_x_kernel(const float* __restrict__ x, float* __restrict__ xt,
                                   _Float16* __restrict__ xt16) {
  int i = blockIdx.x * 256 + threadIdx.x;
  if (i >= BATCH * NPTS) return;
  int b = i >> 11, n = i & (NPTS - 1);
#pragma unroll
  for (int c = 0; c < 3; ++c) {
    float v = x[((size_t)b * NPTS + n) * 3 + c];
    xt[((size_t)b * 3 + c) * NPTS + n] = v;
    xt16[((size_t)b * 3 + c) * NPTS + n] = (_Float16)v;
  }
}

// W (O,Cin) fp32 -> f16 with zero-padded row stride ldw
__global__ void convert_w_kernel(const float* __restrict__ W, _Float16* __restrict__ Wh,
                                 int O, int Cin, int ldw) {
  int i = blockIdx.x * 256 + threadIdx.x;
  if (i >= O * ldw) return;
  int o = i / ldw, c = i - o * ldw;
  Wh[i] = (_Float16)(c < Cin ? W[o * Cin + c] : 0.f);
}

// per-point squared norms of feature slice (C rows of length N, per batch)
__global__ void norms_kernel(const float* __restrict__ hp0, long long bstride,
                             int C, float* __restrict__ nrm) {
  int i = blockIdx.x * 256 + threadIdx.x;
  if (i >= BATCH * NPTS) return;
  int b = i >> 11, m = i & (NPTS - 1);
  const float* hb = hp0 + (size_t)b * bstride;
  float s = 0.f;
  for (int c = 0; c < C; ++c) {
    float v = hb[(size_t)c * NPTS + m];
    s += v * v;
  }
  nrm[i] = s;
}

// ---------------------------------------------------------------------------
// KNN: one wave per query row n; register top-20 per lane, LDS+shuffle merge.
// key = 2*<h_n,h_m> - |h_m|^2  (row-constant |h_n|^2 dropped; order invariant)
// ---------------------------------------------------------------------------
__global__ __launch_bounds__(256) void knn_kernel(const float* __restrict__ hp0,
                                                  long long bstride,
                                                  const float* __restrict__ nrm,
                                                  int C, int* __restrict__ idxout) {
  __shared__ float skey[8][32 * KNN];
  __shared__ int   sid[8][32 * KNN];
  int wv = threadIdx.x >> 5, lane = threadIdx.x & 31;
  int row = blockIdx.x * 8 + wv;
  int b = row >> 11, n = row & (NPTS - 1);
  const float* hb = hp0 + (size_t)b * bstride;

  float best[KNN];
  int   bid[KNN];
#pragma unroll
  for (int j = 0; j < KNN; ++j) { best[j] = -__builtin_inff(); bid[j] = 0; }

  for (int mb = 0; mb < NPTS; mb += 32) {
    int m = mb + lane;
    float dot = 0.f;
    for (int c = 0; c < C; ++c)
      dot += hb[(size_t)c * NPTS + n] * hb[(size_t)c * NPTS + m];
    float kcur = 2.f * dot - nrm[b * NPTS + m];
    int icur = m;
#pragma unroll
    for (int j = 0; j < KNN; ++j) {   // bubble insert, static indices -> VGPRs
      bool gt = kcur > best[j];
      float tf = best[j]; int ti = bid[j];
      if (gt) { best[j] = kcur; bid[j] = icur; kcur = tf; icur = ti; }
    }
  }

#pragma unroll
  for (int j = 0; j < KNN; ++j) {
    skey[wv][lane * KNN + j] = best[j];
    sid[wv][lane * KNN + j]  = bid[j];
  }
  __syncthreads();

  for (int slot = 0; slot < KNN; ++slot) {
    float mk = -__builtin_inff(); int mp = 0;
#pragma unroll
    for (int j = 0; j < KNN; ++j) {
      float v = skey[wv][lane * KNN + j];
      if (v > mk) { mk = v; mp = lane * KNN + j; }
    }
#pragma unroll
    for (int off = 16; off; off >>= 1) {
      float ok = __shfl_xor(mk, off, 32);
      int   op = __shfl_xor(mp, off, 32);
      if (ok > mk || (ok == mk && op < mp)) { mk = ok; mp = op; }
    }
    if (lane == 0) {
      idxout[((size_t)b * NPTS + n) * KNN + slot] = sid[wv][mp];
      skey[wv][mp] = -__builtin_inff();
    }
    __syncthreads();
  }
}

// ---------------------------------------------------------------------------
// EdgeConv GEMM, WMMA 16x16x32 f16, fp32 accum. One block owns 16 points of
// one batch. k-invariant "center" half of e staged once by the TDM from the
// f16 feature mirror; only the C gather-difference rows restage per k.
// MODE 0: accumulate per-channel sum/sum^2 (BN over (B,N,K)).
// MODE 1: BN affine + LeakyReLU, max over k in registers, store f32 + f16 out.
// ---------------------------------------------------------------------------
template <int MODE>
__global__ __launch_bounds__(128) void edgeconv_kernel(
    const float* __restrict__ hp0, long long bstride,
    const _Float16* __restrict__ src16, long long bstride16,
    const int* __restrict__ idx,
    const _Float16* __restrict__ Wh, int ldw,
    const float* __restrict__ bias,
    const float* __restrict__ scale, const float* __restrict__ shiftv,
    float* __restrict__ sumbuf, float* __restrict__ sqbuf,
    float* __restrict__ hout0, _Float16* __restrict__ hout16,
    int C, int O, int chunks) {
  __shared__ _Float16 et[256 * 16];   // [c][pt], row = 16 halves = 32B
  int b = blockIdx.y;
  int ptbase = blockIdx.x * 16;
  int w = threadIdx.x >> 5;
  int lane = threadIdx.x & 31;
  int hh = lane >> 4, nl = lane & 15;
  int R = O >> 4;
  int cw = chunks * 32;
  const float* hb = hp0 + (size_t)b * bstride;

  // async TDM: center rows e[C..2C) = h16[:, ptbase..ptbase+15]
  if (threadIdx.x < 32) {
    unsigned ldsbase = (unsigned)(uintptr_t)(void*)&et[0];
    tdm_load_2d(ldsbase + (unsigned)C * 32u,
                src16 + (size_t)b * bstride16 + ptbase,
                NPTS, (unsigned)C, 16u, (unsigned)C, NPTS);
  }
  // zero the K-padding rows [2C, cw) once
  for (int t = threadIdx.x; t < (cw - 2 * C) * 16; t += 128) et[2 * C * 16 + t] = (_Float16)0.f;
  if (threadIdx.x < 32) __builtin_amdgcn_s_wait_tensorcnt(0);
  __syncthreads();

  v8f sf[4], qf[4], mf[4];
#pragma unroll
  for (int rr = 0; rr < 4; ++rr)
#pragma unroll
    for (int v = 0; v < 8; ++v) {
      sf[rr][v] = 0.f; qf[rr][v] = 0.f; mf[rr][v] = -__builtin_inff();
    }

  for (int k = 0; k < KNN; ++k) {
    // stage difference rows: e[c, pt] = h[c, idx[pt,k]] - h[c, pt]
    for (int t = threadIdx.x; t < C * 16; t += 128) {
      int c = t >> 4, pt = t & 15;
      int ptg = ptbase + pt;
      const float* rowp = hb + (size_t)c * NPTS;
      int nb = idx[((size_t)b * NPTS + ptg) * KNN + k];
      et[t] = (_Float16)(rowp[nb] - rowp[ptg]);
    }
    __syncthreads();

#pragma unroll
    for (int rr = 0; rr < 4; ++rr) {
      int r = w + rr * 4;
      if (r >= R) continue;
      v8f acc = {};
      for (int ch = 0; ch < chunks; ++ch) {
        v16h a  = load_a_frag(Wh, ldw, r * 16, ch * 32);
        v16h bf = load_b_frag16(et, ch * 32);
        acc = __builtin_amdgcn_wmma_f32_16x16x32_f16(false, a, false, bf,
                                                     (short)0, acc, false, false);
      }
#pragma unroll
      for (int v = 0; v < 8; ++v) {
        int o = r * 16 + v + 8 * hh;
        float hv = acc[v] + bias[o];
        if (MODE == 0) {
          sf[rr][v] += hv;
          qf[rr][v] += hv * hv;
        } else {
          hv = hv * scale[o] + shiftv[o];
          hv = (hv >= 0.f) ? hv : 0.2f * hv;
          mf[rr][v] = fmaxf(mf[rr][v], hv);
        }
      }
    }
    __syncthreads();
  }

#pragma unroll
  for (int rr = 0; rr < 4; ++rr) {
    int r = w + rr * 4;
    if (r >= R) continue;
#pragma unroll
    for (int v = 0; v < 8; ++v) {
      int o = r * 16 + v + 8 * hh;
      if (MODE == 0) {
        float s = sf[rr][v], q = qf[rr][v];
#pragma unroll
        for (int off = 1; off < 16; off <<= 1) {
          s += __shfl_xor(s, off, 32);
          q += __shfl_xor(q, off, 32);
        }
        if (nl == 0) {
          atomicAdd(&sumbuf[o], s);
          atomicAdd(&sqbuf[o], q);
        }
      } else {
        size_t oi = (size_t)b * FEATC * NPTS + (size_t)o * NPTS + ptbase + nl;
        hout0[oi]  = mf[rr][v];
        hout16[oi] = (_Float16)mf[rr][v];
      }
    }
  }
}

// scale = g * rsqrt(var+eps); shift = be - mean*scale
__global__ void bn_finalize_kernel(const float* __restrict__ sum, const float* __restrict__ sq,
                                   const float* __restrict__ g, const float* __restrict__ be,
                                   float* __restrict__ scale, float* __restrict__ shiftv,
                                   int O, float invcnt) {
  int o = blockIdx.x * 256 + threadIdx.x;
  if (o >= O) return;
  float m = sum[o] * invcnt;
  float v = sq[o] * invcnt - m * m;
  float s = g[o] * rsqrtf(v + 1e-5f);
  scale[o] = s;
  shiftv[o] = be[o] - m * s;
}

// ---------------------------------------------------------------------------
// Final 1024x512 GEMM over 16-point tiles; whole activation tile staged by one
// TDM descriptor (512x16 f16); max over points fused via shuffle + atomicMax.
// ---------------------------------------------------------------------------
__global__ __launch_bounds__(128) void final_kernel(const _Float16* __restrict__ feat16,
                                                    const _Float16* __restrict__ Wfh,
                                                    const float* __restrict__ bf,
                                                    unsigned* __restrict__ oenc) {
  __shared__ _Float16 et[FEATC * 16];   // [c][pt]
  int b = blockIdx.y;
  int ptbase = blockIdx.x * 16;
  if (threadIdx.x < 32) {
    unsigned ldsbase = (unsigned)(uintptr_t)(void*)&et[0];
    tdm_load_2d(ldsbase, feat16 + (size_t)b * FEATC * NPTS + ptbase,
                NPTS, FEATC, 16u, FEATC, NPTS);
    __builtin_amdgcn_s_wait_tensorcnt(0);
  }
  __syncthreads();
  int w = threadIdx.x >> 5, lane = threadIdx.x & 31;
  int hh = lane >> 4, nl = lane & 15;
  for (int r = w; r < 64; r += 4) {
    v8f acc = {};
#pragma unroll
    for (int ch = 0; ch < 16; ++ch) {
      v16h a  = load_a_frag(Wfh, FEATC, r * 16, ch * 32);
      v16h bb = load_b_frag16(et, ch * 32);
      acc = __builtin_amdgcn_wmma_f32_16x16x32_f16(false, a, false, bb,
                                                   (short)0, acc, false, false);
    }
#pragma unroll
    for (int v = 0; v < 8; ++v) {
      int o = r * 16 + v + 8 * hh;
      float hv = acc[v] + bf[o];
#pragma unroll
      for (int off = 1; off < 16; off <<= 1) hv = fmaxf(hv, __shfl_xor(hv, off, 32));
      if (nl == 0) atomicMax(&oenc[b * 1024 + o], encf(hv));
    }
  }
}

// ---------------------------------------------------------------------------
extern "C" void kernel_launch(void* const* d_in, const int* in_sizes, int n_in,
                              void* d_out, int out_size, void* d_ws, size_t ws_size,
                              hipStream_t stream) {
  (void)in_sizes; (void)n_in; (void)out_size; (void)ws_size;

  const float* x  = (const float*)d_in[0];
  const int* idx0 = (const int*)d_in[1];
  const float* W[4]  = {(const float*)d_in[2],  (const float*)d_in[6],
                        (const float*)d_in[10], (const float*)d_in[14]};
  const float* bb[4] = {(const float*)d_in[3],  (const float*)d_in[7],
                        (const float*)d_in[11], (const float*)d_in[15]};
  const float* gg[4] = {(const float*)d_in[4],  (const float*)d_in[8],
                        (const float*)d_in[12], (const float*)d_in[16]};
  const float* be[4] = {(const float*)d_in[5],  (const float*)d_in[9],
                        (const float*)d_in[13], (const float*)d_in[17]};
  const float* Wf = (const float*)d_in[18];
  const float* bf = (const float*)d_in[19];

  char* p = (char*)d_ws;
  auto carve = [&](size_t bytes) {
    void* r = (void*)p;
    p += (bytes + 255) & ~(size_t)255;
    return r;
  };
  float* xt       = (float*)carve(sizeof(float) * BATCH * 3 * NPTS);
  _Float16* xt16  = (_Float16*)carve(2 * BATCH * 3 * NPTS);
  float* feat     = (float*)carve(sizeof(float) * BATCH * FEATC * NPTS);
  _Float16* feat16 = (_Float16*)carve((size_t)2 * BATCH * FEATC * NPTS);
  _Float16* Wh0 = (_Float16*)carve(2 * 64 * 32);
  _Float16* Wh1 = (_Float16*)carve(2 * 64 * 128);
  _Float16* Wh2 = (_Float16*)carve(2 * 128 * 128);
  _Float16* Wh3 = (_Float16*)carve(2 * 256 * 256);
  _Float16* Wfh = (_Float16*)carve(2 * 1024 * 512);
  float* sumb = (float*)carve(sizeof(float) * 512);
  float* sqb  = (float*)carve(sizeof(float) * 512);
  float* scb  = (float*)carve(sizeof(float) * 512);
  float* shb  = (float*)carve(sizeof(float) * 512);
  int* idxb   = (int*)carve(sizeof(int) * BATCH * NPTS * KNN);
  float* nrm  = (float*)carve(sizeof(float) * BATCH * NPTS);
  unsigned* oenc = (unsigned*)carve(sizeof(unsigned) * BATCH * 1024);
  _Float16* Wh[4] = {Wh0, Wh1, Wh2, Wh3};

  transpose_x_kernel<<<64, 256, 0, stream>>>(x, xt, xt16);
  convert_w_kernel<<<(64 * 32 + 255) / 256, 256, 0, stream>>>(W[0], Wh0, 64, 6, 32);
  convert_w_kernel<<<(64 * 128 + 255) / 256, 256, 0, stream>>>(W[1], Wh1, 64, 128, 128);
  convert_w_kernel<<<(128 * 128 + 255) / 256, 256, 0, stream>>>(W[2], Wh2, 128, 128, 128);
  convert_w_kernel<<<(256 * 256 + 255) / 256, 256, 0, stream>>>(W[3], Wh3, 256, 256, 256);
  convert_w_kernel<<<(1024 * 512 + 255) / 256, 256, 0, stream>>>(Wf, Wfh, 1024, 512, 512);

  const float invcnt = 1.f / (float)((long long)BATCH * NPTS * KNN);
  dim3 cgrid(NPTS / 16, BATCH);

  // ----- layer 0 (input xt, given indices) -----
  zero_kernel<<<2, 256, 0, stream>>>(sumb, 512);
  zero_kernel<<<2, 256, 0, stream>>>(sqb, 512);
  edgeconv_kernel<0><<<cgrid, 128, 0, stream>>>(xt, (long long)3 * NPTS, xt16,
                                                (long long)3 * NPTS, idx0, Wh0, 32,
                                                bb[0], scb, shb, sumb, sqb,
                                                feat, feat16, 3, 64, 1);
  bn_finalize_kernel<<<1, 256, 0, stream>>>(sumb, sqb, gg[0], be[0], scb, shb, 64, invcnt);
  edgeconv_kernel<1><<<cgrid, 128, 0, stream>>>(xt, (long long)3 * NPTS, xt16,
                                                (long long)3 * NPTS, idx0, Wh0, 32,
                                                bb[0], scb, shb, sumb, sqb,
                                                feat, feat16, 3, 64, 1);

  // ----- layers 1..3 (KNN on previous output) -----
  int cinoff[3] = {0, 64, 128};
  int coff[3]   = {64, 128, 256};
  int Cs[3]     = {64, 64, 128};
  int Os[3]     = {64, 128, 256};
  for (int i = 0; i < 3; ++i) {
    const float* hp0      = feat   + (size_t)cinoff[i] * NPTS;
    const _Float16* hp16  = feat16 + (size_t)cinoff[i] * NPTS;
    long long bs = (long long)FEATC * NPTS;
    norms_kernel<<<64, 256, 0, stream>>>(hp0, bs, Cs[i], nrm);
    knn_kernel<<<BATCH * NPTS / 8, 256, 0, stream>>>(hp0, bs, nrm, Cs[i], idxb);
    zero_kernel<<<2, 256, 0, stream>>>(sumb, 512);
    zero_kernel<<<2, 256, 0, stream>>>(sqb, 512);
    int chunks = (2 * Cs[i]) / 32;
    edgeconv_kernel<0><<<cgrid, 128, 0, stream>>>(hp0, bs, hp16, bs, idxb, Wh[i + 1],
                                                  2 * Cs[i], bb[i + 1], scb, shb,
                                                  sumb, sqb, feat, feat16,
                                                  Cs[i], Os[i], chunks);
    bn_finalize_kernel<<<1, 256, 0, stream>>>(sumb, sqb, gg[i + 1], be[i + 1], scb, shb,
                                              Os[i], invcnt);
    edgeconv_kernel<1><<<cgrid, 128, 0, stream>>>(hp0, bs, hp16, bs, idxb, Wh[i + 1],
                                                  2 * Cs[i], bb[i + 1], scb, shb,
                                                  sumb, sqb,
                                                  feat   + (size_t)coff[i] * NPTS,
                                                  feat16 + (size_t)coff[i] * NPTS,
                                                  Cs[i], Os[i], chunks);
  }

  // ----- final linear + global max over points -----
  init_enc_kernel<<<32, 256, 0, stream>>>(oenc, BATCH * 1024);
  final_kernel<<<cgrid, 128, 0, stream>>>(feat16, Wfh, bf, oenc);
  decode_kernel<<<32, 256, 0, stream>>>(oenc, (float*)d_out, BATCH * 1024);
}
